// ExpertConstellation_51410758533301
// MI455X (gfx1250) — matrix operand order repaired and checked by reading, not compiled, not run on hardware
//
#include <hip/hip_runtime.h>
#include <hip/hip_bf16.h>
#include <stdint.h>

// Problem constants (from reference): B=2,S=2048,D=1024,H=2048,O=1024,E=8,TOPK=2
#define NTOK  4096       // B*S
#define D_    1024
#define H_    2048
#define O_    1024
#define E_    8
#define MT    32         // tokens per workgroup tile (two 16-row WMMA A tiles)
#define WAVES 8
#define BLOCK (WAVES * 32)              // 256 threads = 8 waves = 2 per SIMD32
#define KT1   (D_ / 32)  // 32 K-steps, GEMM1
#define NT1   (H_ / 16)  // 128 N-tiles, GEMM1
#define KT2   (H_ / 32)  // 64 K-steps, GEMM2
#define NT2   (O_ / 16)  // 64 N-tiles, GEMM2
#define MTILES ((NTOK + MT - 1) / MT)   // 128 worst-case tiles per expert

typedef __attribute__((ext_vector_type(16))) __bf16        v16bf;
typedef __attribute__((ext_vector_type(8)))  float         v8f;
typedef __attribute__((ext_vector_type(8)))  unsigned int  v8u;

// Workspace layout (bytes)
#define XB_OFF   0ull                          // bf16 x, row-major: 4,194,304 elems
#define W1_OFF   8388608ull                    // bf16 W1, tile-swizzled: 16,777,216 elems
#define W2_OFF   41943040ull                   // bf16 W2, tile-swizzled: 16,777,216 elems
#define CNT_OFF  75497472ull                   // 8 ints
#define LIST_OFF 75497600ull                   // E*4096 ints
#define GATE_OFF 75628672ull                   // E*4096 floats
#define SMEM_BYTES (65536 + 131072 + 128 + 128)

__device__ __forceinline__ unsigned short f2bf(float f) {
  unsigned int u = __builtin_bit_cast(unsigned int, f);
  unsigned int r = (u + 0x7FFFu + ((u >> 16) & 1u)) >> 16;   // RNE
  return (unsigned short)r;
}

__device__ __forceinline__ float gelu_tanh(float x) {
  float x3 = x * x * x;
  return 0.5f * x * (1.0f + tanhf(0.7978845608028654f * (x + 0.044715f * x3)));
}

__device__ __forceinline__ v8f wmma_bf16(v8u a, v8u b, v8f c) {
  return __builtin_amdgcn_wmma_f32_16x16x32_bf16(
      false, __builtin_bit_cast(v16bf, a),
      false, __builtin_bit_cast(v16bf, b),
      (short)0, c, false, false);
}

// ---------------------------------------------------------------- zero fill
__global__ void mk_zero_f32(float* __restrict__ p, int n) {
  for (int i = blockIdx.x * blockDim.x + threadIdx.x; i < n;
       i += gridDim.x * blockDim.x) p[i] = 0.0f;
}
__global__ void mk_zero_i32(int* __restrict__ p, int n) {
  int i = blockIdx.x * blockDim.x + threadIdx.x;
  if (i < n) p[i] = 0;
}

// ------------------------------------------------- f32 -> bf16, row major x
__global__ void mk_cvt_x(const float* __restrict__ x,
                         unsigned short* __restrict__ xb, int n) {
  int i = blockIdx.x * blockDim.x + threadIdx.x;
  if (i < n) xb[i] = f2bf(x[i]);
}

// ---------------- f32 -> bf16 weight conversion into WMMA B-tile swizzle.
// Dst element idx encodes (e, tile=(kt,nt), lane L, elem i). Code's own (contiguous-k)
// B (KxN) 16-bit tile: lane L holds column N = nt*16 + L%16, elements i hold
// K-in-tile = (L/16)*16 + i. GEMM inner loop then loads 32 contiguous bytes.
__global__ void mk_cvt_w(const float* __restrict__ w,
                         unsigned short* __restrict__ wb, int K, int N) {
  long long idx = (long long)blockIdx.x * blockDim.x + threadIdx.x;
  long long tot = (long long)E_ * K * N;
  if (idx >= tot) return;
  int KN = K * N;
  int e  = (int)(idx / KN);
  int r  = (int)(idx % KN);
  int tl = r >> 9;          // tile index (512 elems per 32x16 tile)
  int L  = (r >> 4) & 31;   // lane
  int i  = r & 15;          // element within lane
  int ntn = N >> 4;
  int kt = tl / ntn, nt = tl % ntn;
  int Ki = kt * 32 + ((L >> 4) << 4) + i;
  int Ni = nt * 16 + (L & 15);
  wb[idx] = f2bf(w[((size_t)e * K + Ki) * N + Ni]);
}

// ---------------------------------------------------------------- router
// One wave per token: 8 logits via strided float4 dot products + shfl_xor
// reduce, softmax over E=8, top-2, atomic append to per-expert lists.
__global__ __launch_bounds__(256) void mk_router(
    const float* __restrict__ x, const float* __restrict__ emb,
    int* __restrict__ counts, int* __restrict__ lists,
    float* __restrict__ gates) {
  int wave = threadIdx.x >> 5, lane = threadIdx.x & 31;
  int token = blockIdx.x * 8 + wave;
  if (token >= NTOK) return;
  const float4* xr = (const float4*)(x + (size_t)token * D_);
  float acc[E_];
#pragma unroll
  for (int e = 0; e < E_; ++e) acc[e] = 0.0f;
  for (int it = 0; it < D_ / 128; ++it) {        // 8 iters, 128 floats each
    float4 xv = xr[it * 32 + lane];
#pragma unroll
    for (int e = 0; e < E_; ++e) {
      float4 ev = ((const float4*)(emb + (size_t)e * D_))[it * 32 + lane];
      acc[e] += xv.x * ev.x + xv.y * ev.y + xv.z * ev.z + xv.w * ev.w;
    }
  }
#pragma unroll
  for (int e = 0; e < E_; ++e)
#pragma unroll
    for (int s = 16; s > 0; s >>= 1) acc[e] += __shfl_xor(acc[e], s, 32);
  if (lane == 0) {
    float mx = acc[0];
    for (int e = 1; e < E_; ++e) mx = fmaxf(mx, acc[e]);
    float p[E_], sum = 0.0f;
    for (int e = 0; e < E_; ++e) { p[e] = __expf(acc[e] - mx); sum += p[e]; }
    float inv = 1.0f / sum;
    int i0 = 0;
    for (int e = 1; e < E_; ++e) if (p[e] > p[i0]) i0 = e;
    int i1 = (i0 == 0) ? 1 : 0;
    for (int e = 0; e < E_; ++e) if (e != i0 && p[e] > p[i1]) i1 = e;
    int pos0 = atomicAdd(&counts[i0], 1);
    lists[i0 * NTOK + pos0] = token; gates[i0 * NTOK + pos0] = p[i0] * inv;
    int pos1 = atomicAdd(&counts[i1], 1);
    lists[i1 * NTOK + pos1] = token; gates[i1 * NTOK + pos1] = p[i1] * inv;
  }
}

// --------------------------------------------- fused 2-layer expert GEMM
// One WG = (expert e, tile of MT=32 gathered tokens). 8 waves split N-tiles.
// Phase 1: h = gelu(x·W1[e] + b1[e]) -> LDS (bf16, WMMA A-tile swizzle)
// Phase 2: out += gate * (h·W2[e] + b2[e]) via f32 global atomics.
__global__ __launch_bounds__(BLOCK) void mk_expert_gemm(
    const unsigned short* __restrict__ xb,
    const unsigned short* __restrict__ w1b,
    const unsigned short* __restrict__ w2b,
    const float* __restrict__ b1, const float* __restrict__ b2,
    const int* __restrict__ counts, const int* __restrict__ lists,
    const float* __restrict__ gates, float* __restrict__ out) {
  extern __shared__ char smem[];
  unsigned short* xs  = (unsigned short*)smem;                       // 64 KB: 64 A-tiles
  unsigned short* hs  = (unsigned short*)(smem + 65536);             // 128 KB: 128 A-tiles
  int*            tok = (int*)(smem + 65536 + 131072);
  float*          gts = (float*)(smem + 65536 + 131072 + 128);

  const int e    = blockIdx.x >> 7;       // / MTILES
  const int tile = blockIdx.x & (MTILES - 1);
  const int count = counts[e];
  const int m0 = tile * MT;
  if (m0 >= count) return;

  const int tid  = threadIdx.x;
  const int lane = tid & 31;
  const int wave = tid >> 5;

  if (tid < MT) {
    int gr = m0 + tid; int t = 0; float g = 0.0f;
    if (gr < count) { t = lists[e * NTOK + gr]; g = gates[e * NTOK + gr]; }
    tok[tid] = t; gts[tid] = g;
  }
  __syncthreads();

  // Stage gathered x rows into LDS as WMMA A tiles (lane L row=L%16, two
  // 16-byte K-chunks per lane -> single contiguous 32 B per (tile,lane)).
  for (int idx = tid; idx < 64 * 32; idx += BLOCK) {
    int tl = idx >> 5, L = idx & 31;
    int mhalf = tl >> 5, kt = tl & 31;
    int row = mhalf * 16 + (L & 15);
    const unsigned short* src =
        xb + (size_t)tok[row] * D_ + kt * 32 + (L >> 4) * 16;
    uint4* dst = (uint4*)(xs + tl * 512 + L * 16);
    dst[0] = *(const uint4*)(src);
    dst[1] = *(const uint4*)(src + 8);
  }
  __syncthreads();

  // ---------------- Phase 1: x(32xD) · W1(DxH) + b1 -> gelu -> hs
  const size_t w1base = (size_t)e * D_ * H_;
  for (int j = 0; j < NT1 / WAVES; ++j) {        // 16 N-tiles per wave
    int nt = wave * (NT1 / WAVES) + j;
    v8f acc0 = {}, acc1 = {};
    for (int kt = 0; kt < KT1; ++kt) {
      v8u a0 = *(const v8u*)(xs + kt * 512 + lane * 16);
      v8u a1 = *(const v8u*)(xs + (32 + kt) * 512 + lane * 16);
      const unsigned short* wt =
          w1b + w1base + ((size_t)kt * NT1 + nt) * 512 + lane * 16;
      v8u b = __builtin_nontemporal_load((const v8u*)wt);  // no WGP$ reuse
      __builtin_prefetch(wt + (size_t)NT1 * 512, 0, 0);    // next K-step tile
      acc0 = wmma_bf16(a0, b, acc0);
      acc1 = wmma_bf16(a1, b, acc1);
    }
    // bias + gelu, scatter into hs in A-tile swizzle for phase 2
    int hcol = nt * 16 + (lane & 15);
    float bias = b1[e * H_ + hcol];
    int kt2 = hcol >> 5, kk = hcol & 31;
    int hh = kk >> 4;
    int ii = kk & 15;
    int baseRow = (lane >> 4) * 8;
#pragma unroll
    for (int r = 0; r < 8; ++r) {
      int Lp = (baseRow + r) + 16 * hh;   // dest lane in A-tile
      hs[(size_t)(0 * 64 + kt2) * 512 + Lp * 16 + ii] = f2bf(gelu_tanh(acc0[r] + bias));
      hs[(size_t)(1 * 64 + kt2) * 512 + Lp * 16 + ii] = f2bf(gelu_tanh(acc1[r] + bias));
    }
  }
  __syncthreads();

  // ---------------- Phase 2: h(32xH) · W2(HxO) + b2 -> gate * out
  const size_t w2base = (size_t)e * H_ * O_;
  for (int j = 0; j < NT2 / WAVES; ++j) {        // 8 N-tiles per wave
    int nt = wave * (NT2 / WAVES) + j;
    v8f acc0 = {}, acc1 = {};
    for (int kt = 0; kt < KT2; ++kt) {
      v8u a0 = *(const v8u*)(hs + kt * 512 + lane * 16);
      v8u a1 = *(const v8u*)(hs + (64 + kt) * 512 + lane * 16);
      const unsigned short* wt =
          w2b + w2base + ((size_t)kt * NT2 + nt) * 512 + lane * 16;
      v8u b = __builtin_nontemporal_load((const v8u*)wt);
      __builtin_prefetch(wt + (size_t)NT2 * 512, 0, 0);
      acc0 = wmma_bf16(a0, b, acc0);
      acc1 = wmma_bf16(a1, b, acc1);
    }
    int col = nt * 16 + (lane & 15);
    float bias = b2[e * O_ + col];
    int baseRow = (lane >> 4) * 8;
#pragma unroll
    for (int r = 0; r < 8; ++r) {
      int mr = baseRow + r;
      float v0 = (acc0[r] + bias) * gts[mr];
      float v1 = (acc1[r] + bias) * gts[16 + mr];
      atomicAdd(&out[(size_t)tok[mr] * O_ + col], v0);
      atomicAdd(&out[(size_t)tok[16 + mr] * O_ + col], v1);
    }
  }
}

extern "C" void kernel_launch(void* const* d_in, const int* in_sizes, int n_in,
                              void* d_out, int out_size, void* d_ws, size_t ws_size,
                              hipStream_t stream) {
  (void)in_sizes; (void)n_in; (void)out_size; (void)ws_size;
  const float* x   = (const float*)d_in[0];   // (B,S,D)
  const float* emb = (const float*)d_in[1];   // (E,D)
  const float* W1  = (const float*)d_in[2];   // (E,D,H)
  const float* b1  = (const float*)d_in[3];   // (E,H)
  const float* W2  = (const float*)d_in[4];   // (E,H,O)
  const float* b2  = (const float*)d_in[5];   // (E,O)
  float* out = (float*)d_out;                 // (B,S,O)

  char* ws = (char*)d_ws;
  unsigned short* xb  = (unsigned short*)(ws + XB_OFF);
  unsigned short* w1b = (unsigned short*)(ws + W1_OFF);
  unsigned short* w2b = (unsigned short*)(ws + W2_OFF);
  int*   counts = (int*)(ws + CNT_OFF);
  int*   lists  = (int*)(ws + LIST_OFF);
  float* gatesp = (float*)(ws + GATE_OFF);

  // 1) zero output (atomically accumulated) + routing counters
  mk_zero_f32<<<2048, 256, 0, stream>>>(out, NTOK * O_);
  mk_zero_i32<<<1, 32, 0, stream>>>(counts, E_);

  // 2) precision/layout conversion (bf16; weights pre-swizzled for WMMA B)
  mk_cvt_x<<<(NTOK * D_ + 255) / 256, 256, 0, stream>>>(x, xb, NTOK * D_);
  mk_cvt_w<<<(E_ * D_ * H_ + 255) / 256, 256, 0, stream>>>(W1, w1b, D_, H_);
  mk_cvt_w<<<(E_ * H_ * O_ + 255) / 256, 256, 0, stream>>>(W2, w2b, H_, O_);

  // 3) top-2 routing
  mk_router<<<(NTOK + 7) / 8, 256, 0, stream>>>(x, emb, counts, lists, gatesp);

  // 4) fused expert MLP (needs 192 KB dynamic LDS; CDNA5 WGP has 320 KB)
  (void)hipFuncSetAttribute((const void*)mk_expert_gemm,
                            hipFuncAttributeMaxDynamicSharedMemorySize,
                            SMEM_BYTES);
  mk_expert_gemm<<<E_ * MTILES, BLOCK, SMEM_BYTES, stream>>>(
      xb, w1b, w2b, b1, b2, counts, lists, gatesp, out);
}
